// EncoderLayer_14628658610677
// MI455X (gfx1250) — matrix operand order repaired and checked by reading, not compiled, not run on hardware
//
#include <hip/hip_runtime.h>

// ---------------- problem constants (match reference) ----------------
constexpr int Bb  = 4;
constexpr int Ss  = 1024;
constexpr int Dd  = 1024;
constexpr int Hh  = 16;
constexpr int DKk = 64;
constexpr int Ee  = 16;
constexpr int DHh = 2048;
constexpr int Kk  = 2;
constexpr int Cc  = 1024;
constexpr int Tt  = Bb * Ss;   // 4096 tokens

typedef unsigned int u32;
typedef __attribute__((ext_vector_type(16))) _Float16 v16h;
typedef __attribute__((ext_vector_type(8)))  _Float16 v8h;
typedef __attribute__((ext_vector_type(8)))  float    v8f;

union AF { v16h v; u32 u[8]; };

// CDNA5 async global->LDS copy (ASYNCcnt path, no VGPR data movement).
// dsaddr = LDS byte offset (generic LDS pointer low 32 bits per ISA 10.2),
// vaddr = 64-bit global address, GV mode ("off" = no SADDR).
__device__ __forceinline__ void async_ld_b128(const void* gp, const void* lp) {
  u32 lo = (u32)(size_t)lp;
  asm volatile("global_load_async_to_lds_b128 %0, %1, off"
               :: "v"(lo), "v"(gp) : "memory");
}
__device__ __forceinline__ void async_wait0() {
  asm volatile("s_wait_asynccnt 0x0" ::: "memory");
}

// ---------------- block reductions (wave32) ----------------
__device__ __forceinline__ float block_sum(float v, float* sb) {
  __syncthreads();
  #pragma unroll
  for (int off = 16; off > 0; off >>= 1) v += __shfl_down(v, off, 32);
  int lane = threadIdx.x & 31, w = threadIdx.x >> 5;
  if (lane == 0) sb[w] = v;
  __syncthreads();
  if (w == 0) {
    float t = (lane < (int)(blockDim.x >> 5)) ? sb[lane] : 0.f;
    #pragma unroll
    for (int off = 4; off > 0; off >>= 1) t += __shfl_down(t, off, 32);
    if (lane == 0) sb[0] = t;
  }
  __syncthreads();
  return sb[0];
}

__device__ __forceinline__ float block_max(float v, float* sb) {
  __syncthreads();
  #pragma unroll
  for (int off = 16; off > 0; off >>= 1) v = fmaxf(v, __shfl_down(v, off, 32));
  int lane = threadIdx.x & 31, w = threadIdx.x >> 5;
  if (lane == 0) sb[w] = v;
  __syncthreads();
  if (w == 0) {
    float t = (lane < (int)(blockDim.x >> 5)) ? sb[lane] : -1e30f;
    #pragma unroll
    for (int off = 4; off > 0; off >>= 1) t = fmaxf(t, __shfl_down(t, off, 32));
    if (lane == 0) sb[0] = t;
  }
  __syncthreads();
  return sb[0];
}

// ---------------- generic batched WMMA GEMM (double-buffered, async LDS) ----
// C[M,N](f32 acc) = alpha * A[M,K] x (BT ? B[N,K]^T : B[K,N]) (+biasN)(+addT)(relu)
// A,B are f16; output f32 or f16. Batch index z decomposed as i1 = z / bdiv,
// i2 = z % bdiv with independent strides (handles [B,S,H,DK] head slicing).
struct GemmParams {
  const _Float16* A;
  const _Float16* B;
  const float*    biasN;
  const float*    addT;
  void*           Cq;
  int M, N, Kd, lda, ldb, ldc, ldadd, bdiv;
  long sA1, sA2, sB1, sB2, sC1, sC2, sAdd1, sAdd2, sBias1;
  float alpha;
};

template<int BM, int BN, int WR, int WC, int MT, int NT,
         bool BT, bool BIAS, bool ADD, bool RELU, bool F16OUT>
__global__ __launch_bounds__(256) void gemm_wmma_kernel(GemmParams p) {
  constexpr int BK   = 32;
  constexpr int LDAS = 40;   // halves; 80B rows -> 16B aligned, conflict-free strides
  constexpr int LDBS = 40;
  __shared__ _Float16 As[2][BM * LDAS];
  __shared__ _Float16 Bs[2][BN * LDBS];   // n-major: Bs[n][k]

  const int tid  = threadIdx.x;
  const int lane = tid & 31;
  const int wave = tid >> 5;
  const int wr   = wave / WC;
  const int wc   = wave % WC;
  const int mlan = lane & 15;
  const int hgrp = lane >> 4;

  const int bz = blockIdx.z;
  const int i1 = bz / p.bdiv;
  const int i2 = bz % p.bdiv;

  const _Float16* __restrict__ Ag = p.A + i1 * p.sA1 + i2 * p.sA2;
  const _Float16* __restrict__ Bg = p.B + i1 * p.sB1 + i2 * p.sB2;

  const long m0 = (long)blockIdx.y * BM;
  const long n0 = (long)blockIdx.x * BN;

  // issue async loads of one k-slab into LDS buffer `buf`
  auto stageA = [&](int buf, int k0) {
    _Float16* dst = &As[buf][0];
    #pragma unroll
    for (int c = tid; c < BM * BK / 8; c += 256) {
      int r  = c >> 2;            // BK/8 == 4 chunks per row
      int kc = (c & 3) * 8;
      async_ld_b128(Ag + (m0 + r) * (long)p.lda + k0 + kc, dst + r * LDAS + kc);
    }
  };
  auto stageB = [&](int buf, int k0) {
    _Float16* dst = &Bs[buf][0];
    if constexpr (BT) {
      #pragma unroll
      for (int c = tid; c < BN * BK / 8; c += 256) {
        int n  = c >> 2;
        int kc = (c & 3) * 8;
        async_ld_b128(Bg + (n0 + n) * (long)p.ldb + k0 + kc, dst + n * LDBS + kc);
      }
    } else {
      constexpr int CPR = BN / 8;
      #pragma unroll
      for (int c = tid; c < BN * BK / 8; c += 256) {
        int k  = c / CPR;
        int nc = (c % CPR) * 8;
        v8h val = *(const v8h*)(Bg + (long)(k0 + k) * p.ldb + n0 + nc);
        #pragma unroll
        for (int j = 0; j < 8; ++j) dst[(nc + j) * LDBS + k] = val[j];
      }
    }
  };

  const v8f zero8 = {0.f,0.f,0.f,0.f,0.f,0.f,0.f,0.f};
  v8f acc[MT][NT];
  #pragma unroll
  for (int i = 0; i < MT; ++i)
    #pragma unroll
    for (int j = 0; j < NT; ++j) acc[i][j] = zero8;

  const int nk = p.Kd / BK;
  stageA(0, 0);
  stageB(0, 0);

  for (int i = 0; i < nk; ++i) {
    async_wait0();        // this wave's pending async fills complete
    __syncthreads();      // all waves' fills of buffer (i&1) visible
    const int cur = i & 1;

    if (i + 1 < nk) {
      const int kn = (i + 1) * BK;
      if (i + 2 < nk) {   // L2 prefetch two slabs ahead
        __builtin_prefetch(Ag + (m0 + (tid & (BM - 1))) * (long)p.lda + kn + BK, 0, 1);
        if (BT) __builtin_prefetch(Bg + (n0 + (tid & (BN - 1))) * (long)p.ldb + kn + BK, 0, 1);
        else    __builtin_prefetch(Bg + (long)(kn + BK + (tid & (BK - 1))) * p.ldb + n0, 0, 1);
      }
      stageA(cur ^ 1, kn);  // async: overlaps with WMMAs below
      stageB(cur ^ 1, kn);
    }

    // ---- gather fragments per ISA 7.12.2 layouts, issue WMMAs ----
    AF af[MT];
    AF bf[NT];
    #pragma unroll
    for (int mt = 0; mt < MT; ++mt) {
      const u32* arow = (const u32*)(&As[cur][0] + (wr * MT * 16 + mt * 16 + mlan) * LDAS);
      #pragma unroll
      for (int i8 = 0; i8 < 8; ++i8) {
        int idx = ((i8 < 4) ? 0 : 8) + hgrp * 4 + (i8 & 3);
        af[mt].u[i8] = arow[idx];
      }
    }
    #pragma unroll
    for (int nt = 0; nt < NT; ++nt) {
      const u32* brow = (const u32*)(&Bs[cur][0] + (wc * NT * 16 + nt * 16 + mlan) * LDBS);
      #pragma unroll
      for (int i8 = 0; i8 < 8; ++i8) bf[nt].u[i8] = brow[((i8 < 4) ? 0 : 8) + hgrp * 4 + (i8 & 3)];
    }
    #pragma unroll
    for (int mt = 0; mt < MT; ++mt)
      #pragma unroll
      for (int nt = 0; nt < NT; ++nt)
        acc[mt][nt] = __builtin_amdgcn_wmma_f32_16x16x32_f16(
            false, af[mt].v, false, bf[nt].v, (short)0, acc[mt][nt], false, false);
  }

  // ---- epilogue ----
  const float* biasp = BIAS ? (p.biasN + i1 * p.sBias1) : nullptr;
  const float* addp  = ADD  ? (p.addT  + i1 * p.sAdd1 + i2 * p.sAdd2) : nullptr;
  const long coff = i1 * p.sC1 + i2 * p.sC2;
  #pragma unroll
  for (int nt = 0; nt < NT; ++nt) {
    long n = n0 + wc * NT * 16 + nt * 16 + mlan;
    float bn = BIAS ? biasp[n] : 0.f;
    #pragma unroll
    for (int mt = 0; mt < MT; ++mt) {
      #pragma unroll
      for (int r = 0; r < 8; ++r) {
        long m = m0 + wr * MT * 16 + mt * 16 + r + 8 * hgrp;
        float v = acc[mt][nt][r] * p.alpha + bn;
        if (ADD)  v += addp[m * (long)p.ldadd + n];
        if (RELU) v = fmaxf(v, 0.f);
        if (F16OUT) ((_Float16*)p.Cq)[coff + m * (long)p.ldc + n] = (_Float16)v;
        else        ((float*)p.Cq)[coff + m * (long)p.ldc + n] = v;
      }
    }
  }
}

// ---------------- elementwise / normalization kernels ----------------
__global__ __launch_bounds__(256) void f32_to_f16_kernel(const float* __restrict__ in,
                                                         _Float16* __restrict__ out, long n) {
  long i = ((long)blockIdx.x * blockDim.x + threadIdx.x) * 4;
  if (i + 3 < n) {
    float4 v = *(const float4*)(in + i);
    out[i + 0] = (_Float16)v.x; out[i + 1] = (_Float16)v.y;
    out[i + 2] = (_Float16)v.z; out[i + 3] = (_Float16)v.w;
  }
}

__global__ __launch_bounds__(256) void fill_zero_kernel(u32* __restrict__ p, long n) {
  long i = (long)blockIdx.x * blockDim.x + threadIdx.x;
  if (i < n) p[i] = 0u;
}

__global__ __launch_bounds__(256) void ln_kernel(const float* __restrict__ x,
                                                 const float* __restrict__ g,
                                                 const float* __restrict__ b,
                                                 _Float16* __restrict__ out, int D) {
  __shared__ float sb[32];
  long row = blockIdx.x;
  const float* xr = x + row * (long)D;
  float s = 0.f, s2 = 0.f;
  for (int d = threadIdx.x; d < D; d += blockDim.x) { float v = xr[d]; s += v; s2 += v * v; }
  s  = block_sum(s,  sb);
  s2 = block_sum(s2, sb);
  float mu = s / D;
  float var = s2 / D - mu * mu;
  float rs = rsqrtf(var + 1e-5f);
  _Float16* orow = out + row * (long)D;
  for (int d = threadIdx.x; d < D; d += blockDim.x)
    orow[d] = (_Float16)((xr[d] - mu) * rs * g[d] + b[d]);
}

__global__ __launch_bounds__(256) void softmax_rows_kernel(_Float16* __restrict__ sc, int n) {
  __shared__ float sb[32];
  long row = blockIdx.x;
  _Float16* p = sc + row * (long)n;
  float mx = -1e30f;
  for (int i = threadIdx.x; i < n; i += blockDim.x) mx = fmaxf(mx, (float)p[i]);
  mx = block_max(mx, sb);
  float sum = 0.f;
  for (int i = threadIdx.x; i < n; i += blockDim.x) sum += __expf((float)p[i] - mx);
  sum = block_sum(sum, sb);
  float inv = 1.f / sum;
  for (int i = threadIdx.x; i < n; i += blockDim.x)
    p[i] = (_Float16)(__expf((float)p[i] - mx) * inv);
}

// ---------------- MoE routing kernels ----------------
__global__ __launch_bounds__(256) void gate_kernel(const _Float16* __restrict__ zh,
                                                   const float* __restrict__ gw,
                                                   const float* __restrict__ gb,
                                                   float* __restrict__ logits) {
  int t = blockIdx.x * 16 + (threadIdx.x >> 4);
  int e = threadIdx.x & 15;
  const _Float16* zr = zh + (long)t * Dd;
  float acc = 0.f;
  for (int d = 0; d < Dd; ++d) acc += (float)zr[d] * gw[d * Ee + e];
  logits[(long)t * Ee + e] = acc + gb[e];
}

__global__ __launch_bounds__(256) void topk_kernel(const float* __restrict__ logits,
                                                   float* __restrict__ gsc,
                                                   int* __restrict__ e_p) {
  int t = blockIdx.x * blockDim.x + threadIdx.x;
  if (t >= Tt) return;
  const float* lr = logits + (long)t * Ee;
  float v0 = -1e30f, v1 = -1e30f; int i0 = 0, i1 = 0;
  for (int e = 0; e < Ee; ++e) {
    float v = lr[e];
    if (v > v0) { v1 = v0; i1 = i0; v0 = v; i0 = e; }
    else if (v > v1) { v1 = v; i1 = e; }
  }
  float e1 = __expf(v1 - v0);
  float denom = 1.f + e1;
  gsc[t * 2 + 0] = 1.f / denom;
  gsc[t * 2 + 1] = e1 / denom;
  e_p[t * 2 + 0] = i0;
  e_p[t * 2 + 1] = i1;
}

// one block per expert: ordered (token-order) slot assignment via block scan
__global__ __launch_bounds__(256) void pos_scan_kernel(const int* __restrict__ e_p,
                                                       int* __restrict__ pos_p, int P) {
  int e = blockIdx.x;
  __shared__ int sd[256];
  int carry = 0;
  for (int base = 0; base < P; base += 256) {
    int i = base + threadIdx.x;
    int pred = (e_p[i] == e) ? 1 : 0;
    sd[threadIdx.x] = pred;
    __syncthreads();
    for (int off = 1; off < 256; off <<= 1) {
      int t = (threadIdx.x >= off) ? sd[threadIdx.x - off] : 0;
      __syncthreads();
      sd[threadIdx.x] += t;
      __syncthreads();
    }
    if (pred) pos_p[i] = carry + sd[threadIdx.x] - 1;
    carry += sd[255];
    __syncthreads();
  }
}

__global__ __launch_bounds__(256) void dispatch_kernel(const _Float16* __restrict__ zh,
                                                       const int* __restrict__ e_p,
                                                       const int* __restrict__ pos_p,
                                                       _Float16* __restrict__ bufh) {
  int pr = blockIdx.x;
  int e = e_p[pr], pos = pos_p[pr];
  if (pos >= Cc) return;
  int t = pr >> 1;
  const u32* src = (const u32*)(zh + (long)t * Dd);
  u32* dst = (u32*)(bufh + ((long)e * Cc + pos) * (long)Dd);
  for (int i = threadIdx.x; i < Dd / 2; i += 256) dst[i] = src[i];
}

__global__ __launch_bounds__(256) void combine_kernel(const float* __restrict__ x1,
                                                      const float* __restrict__ gsc,
                                                      const int* __restrict__ e_p,
                                                      const int* __restrict__ pos_p,
                                                      const _Float16* __restrict__ ybh,
                                                      float* __restrict__ out) {
  int t = blockIdx.x;
  int e0 = e_p[2 * t], p0 = pos_p[2 * t];
  int e1 = e_p[2 * t + 1], p1 = pos_p[2 * t + 1];
  float g0 = gsc[2 * t], g1 = gsc[2 * t + 1];
  bool ok0 = p0 < Cc, ok1 = p1 < Cc;
  const _Float16* y0 = ybh + ((long)e0 * Cc + (ok0 ? p0 : 0)) * (long)Dd;
  const _Float16* y1 = ybh + ((long)e1 * Cc + (ok1 ? p1 : 0)) * (long)Dd;
  for (int d = threadIdx.x; d < Dd; d += 256) {
    float r = x1[(long)t * Dd + d];
    if (ok0) r += g0 * (float)y0[d];
    if (ok1) r += g1 * (float)y1[d];
    out[(long)t * Dd + d] = r;
  }
}

// ---------------- host orchestration ----------------
extern "C" void kernel_launch(void* const* d_in, const int* in_sizes, int n_in,
                              void* d_out, int out_size, void* d_ws, size_t ws_size,
                              hipStream_t stream) {
  (void)in_sizes; (void)n_in; (void)out_size; (void)ws_size;
  const float* x    = (const float*)d_in[0];
  const float* ab   = (const float*)d_in[1];
  const float* ln1g = (const float*)d_in[2];
  const float* ln1b = (const float*)d_in[3];
  const float* wq   = (const float*)d_in[4];
  const float* bq   = (const float*)d_in[5];
  const float* wk   = (const float*)d_in[6];
  const float* bk   = (const float*)d_in[7];
  const float* wv   = (const float*)d_in[8];
  const float* bv   = (const float*)d_in[9];
  const float* wo   = (const float*)d_in[10];
  const float* bo   = (const float*)d_in[11];
  const float* ln2g = (const float*)d_in[12];
  const float* ln2b = (const float*)d_in[13];
  const float* gw   = (const float*)d_in[14];
  const float* gb   = (const float*)d_in[15];
  const float* w1   = (const float*)d_in[16];
  const float* b1   = (const float*)d_in[17];
  const float* w2   = (const float*)d_in[18];
  const float* b2   = (const float*)d_in[19];

  char* wp = (char*)d_ws;
  auto take = [&](size_t bytes) -> void* {
    void* r = (void*)wp;
    wp += (bytes + 255) & ~(size_t)255;
    return r;
  };
  _Float16* wqh  = (_Float16*)take((size_t)Dd * Dd * 2);
  _Float16* wkh  = (_Float16*)take((size_t)Dd * Dd * 2);
  _Float16* wvh  = (_Float16*)take((size_t)Dd * Dd * 2);
  _Float16* woh  = (_Float16*)take((size_t)Dd * Dd * 2);
  _Float16* w1h  = (_Float16*)take((size_t)Ee * Dd * DHh * 2);
  _Float16* w2h  = (_Float16*)take((size_t)Ee * DHh * Dd * 2);
  _Float16* yh   = (_Float16*)take((size_t)Tt * Dd * 2);
  _Float16* qh   = (_Float16*)take((size_t)Tt * Dd * 2);
  _Float16* kh   = (_Float16*)take((size_t)Tt * Dd * 2);
  _Float16* vh   = (_Float16*)take((size_t)Tt * Dd * 2);
  _Float16* sc   = (_Float16*)take((size_t)Bb * Hh * Ss * Ss * 2);
  _Float16* ao   = (_Float16*)take((size_t)Tt * Dd * 2);
  float*    x1   = (float*)   take((size_t)Tt * Dd * 4);
  _Float16* zh   = (_Float16*)take((size_t)Tt * Dd * 2);
  float*    lg   = (float*)   take((size_t)Tt * Ee * 4);
  float*    gsc  = (float*)   take((size_t)Tt * Kk * 4);
  int*      e_p  = (int*)     take((size_t)Tt * Kk * 4);
  int*      posp = (int*)     take((size_t)Tt * Kk * 4);
  _Float16* bufh = (_Float16*)take((size_t)Ee * Cc * Dd * 2);
  _Float16* hbuf = (_Float16*)take((size_t)Ee * Cc * DHh * 2);
  _Float16* ybh  = (_Float16*)take((size_t)Ee * Cc * Dd * 2);

  auto conv = [&](const float* src, _Float16* dst, long n) {
    long blocks = (n / 4 + 255) / 256;
    f32_to_f16_kernel<<<(unsigned)blocks, 256, 0, stream>>>(src, dst, n);
  };
  conv(wq, wqh, (long)Dd * Dd);
  conv(wk, wkh, (long)Dd * Dd);
  conv(wv, wvh, (long)Dd * Dd);
  conv(wo, woh, (long)Dd * Dd);
  conv(w1, w1h, (long)Ee * Dd * DHh);
  conv(w2, w2h, (long)Ee * DHh * Dd);

  // ---- LN1 ----
  ln_kernel<<<Tt, 256, 0, stream>>>(x, ln1g, ln1b, yh, Dd);

  // ---- QKV projections (WMMA) ----
  {
    GemmParams g{};
    g.M = Tt; g.N = Dd; g.Kd = Dd; g.lda = Dd; g.ldb = Dd; g.ldc = Dd;
    g.bdiv = 1; g.alpha = 1.f; g.A = yh;
    dim3 grid(Dd / 128, Tt / 128, 1);
    g.B = wqh; g.biasN = bq; g.Cq = qh;
    gemm_wmma_kernel<128,128,2,4,4,2,false,true,false,false,true><<<grid,256,0,stream>>>(g);
    g.B = wkh; g.biasN = bk; g.Cq = kh;
    gemm_wmma_kernel<128,128,2,4,4,2,false,true,false,false,true><<<grid,256,0,stream>>>(g);
    g.B = wvh; g.biasN = bv; g.Cq = vh;
    gemm_wmma_kernel<128,128,2,4,4,2,false,true,false,false,true><<<grid,256,0,stream>>>(g);
  }

  // ---- scores = scale * Q K^T + attn_bias  (batched over B*H, BT mode) ----
  {
    GemmParams g{};
    g.M = Ss; g.N = Ss; g.Kd = DKk; g.lda = Dd; g.ldb = Dd; g.ldc = Ss; g.ldadd = Ss;
    g.bdiv = Hh; g.alpha = 0.125f;  // 64^-0.5
    g.A = qh; g.sA1 = (long)Ss * Dd; g.sA2 = DKk;
    g.B = kh; g.sB1 = (long)Ss * Dd; g.sB2 = DKk;
    g.Cq = sc;  g.sC1 = (long)Hh * Ss * Ss; g.sC2 = (long)Ss * Ss;
    g.addT = ab; g.sAdd1 = (long)Hh * Ss * Ss; g.sAdd2 = (long)Ss * Ss;
    dim3 grid(Ss / 128, Ss / 128, Bb * Hh);
    gemm_wmma_kernel<128,128,2,4,4,2,true,false,true,false,true><<<grid,256,0,stream>>>(g);
  }

  // ---- softmax rows (in place on f16 scores) ----
  softmax_rows_kernel<<<Bb * Hh * Ss, 256, 0, stream>>>(sc, Ss);

  // ---- O = P V (batched, N = 64) ----
  {
    GemmParams g{};
    g.M = Ss; g.N = DKk; g.Kd = Ss; g.lda = Ss; g.ldb = Dd; g.ldc = Dd;
    g.bdiv = Hh; g.alpha = 1.f;
    g.A = sc; g.sA1 = (long)Hh * Ss * Ss; g.sA2 = (long)Ss * Ss;
    g.B = vh; g.sB1 = (long)Ss * Dd; g.sB2 = DKk;
    g.Cq = ao; g.sC1 = (long)Ss * Dd; g.sC2 = DKk;
    dim3 grid(DKk / 64, Ss / 128, Bb * Hh);
    gemm_wmma_kernel<128,64,4,2,2,2,false,false,false,false,true><<<grid,256,0,stream>>>(g);
  }

  // ---- x1 = O @ wo + bo + x (residual, f32 out) ----
  {
    GemmParams g{};
    g.M = Tt; g.N = Dd; g.Kd = Dd; g.lda = Dd; g.ldb = Dd; g.ldc = Dd; g.ldadd = Dd;
    g.bdiv = 1; g.alpha = 1.f;
    g.A = ao; g.B = woh; g.biasN = bo; g.addT = x; g.Cq = x1;
    dim3 grid(Dd / 128, Tt / 128, 1);
    gemm_wmma_kernel<128,128,2,4,4,2,false,true,true,false,false><<<grid,256,0,stream>>>(g);
  }

  // ---- LN2 -> zh (f16) ----
  ln_kernel<<<Tt, 256, 0, stream>>>(x1, ln2g, ln2b, zh, Dd);

  // ---- gate logits, top-2, ordered slot assignment ----
  gate_kernel<<<Tt / 16, 256, 0, stream>>>(zh, gw, gb, lg);
  topk_kernel<<<Tt / 256, 256, 0, stream>>>(lg, gsc, e_p);
  pos_scan_kernel<<<Ee, 256, 0, stream>>>(e_p, posp, Tt * Kk);

  // ---- dispatch into zeroed expert buffer ----
  {
    long nuints = (long)Ee * Cc * Dd / 2;
    fill_zero_kernel<<<(unsigned)((nuints + 255) / 256), 256, 0, stream>>>((u32*)bufh, nuints);
  }
  dispatch_kernel<<<Tt * Kk, 256, 0, stream>>>(zh, e_p, posp, bufh);

  // ---- expert GEMM 1: h = relu(buf @ w1 + b1)  (batched over experts) ----
  {
    GemmParams g{};
    g.M = Cc; g.N = DHh; g.Kd = Dd; g.lda = Dd; g.ldb = DHh; g.ldc = DHh;
    g.bdiv = 1; g.alpha = 1.f;
    g.A = bufh; g.sA1 = (long)Cc * Dd;
    g.B = w1h;  g.sB1 = (long)Dd * DHh;
    g.biasN = b1; g.sBias1 = DHh;
    g.Cq = hbuf; g.sC1 = (long)Cc * DHh;
    dim3 grid(DHh / 128, Cc / 128, Ee);
    gemm_wmma_kernel<128,128,2,4,4,2,false,true,false,true,true><<<grid,256,0,stream>>>(g);
  }
  // ---- expert GEMM 2: yb = h @ w2 + b2 ----
  {
    GemmParams g{};
    g.M = Cc; g.N = Dd; g.Kd = DHh; g.lda = DHh; g.ldb = Dd; g.ldc = Dd;
    g.bdiv = 1; g.alpha = 1.f;
    g.A = hbuf; g.sA1 = (long)Cc * DHh;
    g.B = w2h;  g.sB1 = (long)DHh * Dd;
    g.biasN = b2; g.sBias1 = Dd;
    g.Cq = ybh; g.sC1 = (long)Cc * Dd;
    dim3 grid(Dd / 128, Cc / 128, Ee);
    gemm_wmma_kernel<128,128,2,4,4,2,false,true,false,false,true><<<grid,256,0,stream>>>(g);
  }

  // ---- combine: out = x1 + sum_k gate_k * y_k ----
  combine_kernel<<<Tt, 256, 0, stream>>>(x1, gsc, e_p, posp, ybh, (float*)d_out);
}
